// Seq2SeqLSTM_54305566490966
// MI455X (gfx1250) — matrix operand order repaired and checked by reading, not compiled, not run on hardware
//
#include <hip/hip_runtime.h>
#include <hip/hip_bf16.h>
#include <math.h>

// ---------------------------------------------------------------------------
// CDNA5 (gfx1250) wave32 WMMA seq2seq LSTM, v3.
//  - All GEMM operands bf16 (weights pre-converted once; activations native).
//  - Async DMA staging: global_load_async_to_lds_b128 + s_wait_asynccnt,
//    double-buffered so the DMA of tile k+1 overlaps WMMA on tile k.
// ---------------------------------------------------------------------------

typedef __attribute__((ext_vector_type(16))) __bf16 bf16x16;
typedef __attribute__((ext_vector_type(8)))  __bf16 bf16x8;
typedef __attribute__((ext_vector_type(4)))  __bf16 bf16x4;
typedef __attribute__((ext_vector_type(8)))  float  floatx8;

#define WMMA_BF16(a, b, c) \
  __builtin_amdgcn_wmma_f32_16x16x32_bf16(false, (a), false, (b), (short)0, (c), false, false)

// Async global->LDS 16B copy (per-lane addresses), tracked by ASYNCcnt.
__device__ __forceinline__ void async_b128(void* lds, const void* g) {
  unsigned loff = (unsigned)(unsigned long long)
      (__attribute__((address_space(3))) char*)lds;
  asm volatile("global_load_async_to_lds_b128 %0, %1, off"
               :: "v"(loff), "v"((unsigned long long)g) : "memory");
}
__device__ __forceinline__ void async_wait0() {
  asm volatile("s_wait_asynccnt 0x0" ::: "memory");
}

// A fragment: 16x32 bf16, contiguous k order (k = 16*(lane/16) + s), same as frag_b.
__device__ __forceinline__ bf16x16 frag_a(const __bf16* As, int m0, int lane) {
  bf16x16 a;
  const __bf16* p = As + (m0 + (lane & 15)) * 32;
  int kb = (lane < 16) ? 0 : 16;
#pragma unroll
  for (int j = 0; j < 8; ++j) { a[j] = p[kb + j]; a[8 + j] = p[kb + 8 + j]; }
  return a;
}

// B fragment: 32x16 (KxN), Ws stored [n][k].
__device__ __forceinline__ bf16x16 frag_b(const __bf16* Ws, int n0, int lane) {
  bf16x16 b;
  const __bf16* p = Ws + (n0 + (lane & 15)) * 32 + ((lane < 16) ? 0 : 16);
#pragma unroll
  for (int j = 0; j < 16; ++j) b[j] = p[j];
  return b;
}

__device__ __forceinline__ float sigf(float x) { return 1.0f / (1.0f + __expf(-x)); }

// ---------------------------------------------------------------------------
// GEMM: Y[M,N](f32) = A[M,K](bf16) * W[N,K](bf16)^T + bias[N](f32)
// Tile 64x128, BK=32, double-buffered async DMA. Grid: (N/128, M/64).
// ---------------------------------------------------------------------------
__global__ __launch_bounds__(256) void gemm_bf16_wmma(
    const __bf16* __restrict__ A, int lda,
    const __bf16* __restrict__ W, int ldw,
    const float* __restrict__ bias,
    float* __restrict__ Y, int ldy, int K)
{
  __shared__ __attribute__((aligned(32))) __bf16 As[2][64 * 32];
  __shared__ __attribute__((aligned(32))) __bf16 Ws[2][128 * 32];
  const int bn    = blockIdx.x * 128;
  const int bm    = blockIdx.y * 64;
  const int tid   = threadIdx.x;
  const int lane  = tid & 31;
  const int wave  = tid >> 5;
  const int msub  = (wave & 3) * 16;
  const int nbase = (wave >> 2) * 64;

  const int ar = tid >> 2, ak = (tid & 3) * 8;    // A: 8 bf16 per thread
  const int wr = tid >> 1, wk = (tid & 1) * 16;   // W: 16 bf16 per thread

  floatx8 acc[4] = {};

  auto stage = [&](int buf, int k0) {
    async_b128(&As[buf][ar * 32 + ak], &A[(long long)(bm + ar) * lda + k0 + ak]);
    async_b128(&Ws[buf][wr * 32 + wk], &W[(long long)(bn + wr) * ldw + k0 + wk]);
    async_b128(&Ws[buf][wr * 32 + wk + 8], &W[(long long)(bn + wr) * ldw + k0 + wk + 8]);
  };

  stage(0, 0);
  for (int k0 = 0; k0 < K; k0 += 32) {
    const int buf = (k0 >> 5) & 1;
    async_wait0();       // this wave's DMA for `buf` complete
    __syncthreads();     // all waves' DMA visible; prev compute on buf^1 done
    if (k0 + 32 < K) stage(buf ^ 1, k0 + 32);   // overlap DMA with WMMA
    bf16x16 af = frag_a(As[buf], msub, lane);
#pragma unroll
    for (int q = 0; q < 4; ++q) {
      bf16x16 bf = frag_b(Ws[buf], nbase + q * 16, lane);
      acc[q] = WMMA_BF16(af, bf, acc[q]);
    }
  }

  // C/D layout: VGPR r -> M = r + (lane>=16 ? 8 : 0), N = lane & 15
  const int rowBase = bm + msub + ((lane >= 16) ? 8 : 0);
#pragma unroll
  for (int q = 0; q < 4; ++q) {
    const int col = bn + nbase + q * 16 + (lane & 15);
    const float bv = bias[col];
#pragma unroll
    for (int r = 0; r < 8; ++r) {
      Y[(long long)(rowBase + r) * ldy + col] = acc[q][r] + bv;
    }
  }
}

// ---------------------------------------------------------------------------
// Fused LSTM step: G = xg_t + h_in @ Whh^T (bf16 WMMA), gates+cell in-register.
// 8 blocks (64-wide d-slice), 256 thr. Wave w holds ALL 4 gates of its
// (m-strip, d-half) tile. h: bf16 ping-pong. c: f32 block-private.
// Double-buffered async DMA staging, same pipeline as the GEMM.
// ---------------------------------------------------------------------------
__global__ __launch_bounds__(256) void lstm_step_wmma(
    const float* __restrict__ xg_t, long long xg_bstride,
    const __bf16* __restrict__ Whh,          // [2048,512] bf16
    const __bf16* __restrict__ h_in,  int h_in_stride,
    __bf16*       __restrict__ h_out, int h_out_stride,
    float*        __restrict__ c,     int c_stride,
    __bf16*       __restrict__ ys_t,  long long ys_bstride)  // may be null
{
  __shared__ __attribute__((aligned(32))) __bf16 Ah[2][64 * 32];
  __shared__ __attribute__((aligned(32))) __bf16 Bw[2][4][64 * 32];
  const int d0    = blockIdx.x * 64;
  const int tid   = threadIdx.x;
  const int lane  = tid & 31;
  const int wave  = tid >> 5;
  const int msub  = (wave & 3) * 16;
  const int dhalf = (wave >> 2) * 32;

  const int ar = tid >> 2, ak = (tid & 3) * 8;

  floatx8 acc[4][2] = {};

  auto stage = [&](int buf, int k0) {
    async_b128(&Ah[buf][ar * 32 + ak],
               &h_in[(long long)ar * h_in_stride + k0 + ak]);
#pragma unroll
    for (int g = 0; g < 4; ++g) {
      async_b128(&Bw[buf][g][ar * 32 + ak],
                 &Whh[(long long)(g * 512 + d0 + ar) * 512 + k0 + ak]);
    }
  };

  stage(0, 0);
  for (int k0 = 0; k0 < 512; k0 += 32) {
    const int buf = (k0 >> 5) & 1;
    async_wait0();
    __syncthreads();
    if (k0 + 32 < 512) stage(buf ^ 1, k0 + 32);
    bf16x16 af = frag_a(Ah[buf], msub, lane);
#pragma unroll
    for (int g = 0; g < 4; ++g) {
      bf16x16 b0 = frag_b(Bw[buf][g], dhalf, lane);
      bf16x16 b1 = frag_b(Bw[buf][g], dhalf + 16, lane);
      acc[g][0] = WMMA_BF16(af, b0, acc[g][0]);
      acc[g][1] = WMMA_BF16(af, b1, acc[g][1]);
    }
  }

  const int mBase = msub + ((lane >= 16) ? 8 : 0);
#pragma unroll
  for (int ns = 0; ns < 2; ++ns) {
    const int d = d0 + dhalf + ns * 16 + (lane & 15);
#pragma unroll
    for (int r = 0; r < 8; ++r) {
      const int b = mBase + r;
      const long long xb = (long long)b * xg_bstride;
      float i_ = sigf (acc[0][ns][r] + xg_t[xb + d]);
      float f_ = sigf (acc[1][ns][r] + xg_t[xb + 512 + d]);
      float g_ = tanhf(acc[2][ns][r] + xg_t[xb + 1024 + d]);
      float o_ = sigf (acc[3][ns][r] + xg_t[xb + 1536 + d]);
      float cv = f_ * c[(long long)b * c_stride + d] + i_ * g_;
      float hv = o_ * tanhf(cv);
      c[(long long)b * c_stride + d]          = cv;
      h_out[(long long)b * h_out_stride + d]  = (__bf16)hv;
      if (ys_t) ys_t[(long long)b * ys_bstride + d] = (__bf16)hv;
    }
  }
}

// ---------------------------------------------------------------------------
// Embedding gather -> bf16 rows of 512.
// ---------------------------------------------------------------------------
__global__ __launch_bounds__(128) void embed_gather_bf16(
    const int* __restrict__ tok, int tok_ld,
    const float* __restrict__ emb,
    __bf16* __restrict__ out, int T)
{
  int r = blockIdx.x;
  int b = r / T, t = r - b * T;
  int v = tok[b * tok_ld + t];
  float4 f = ((const float4*)(emb + (long long)v * 512))[threadIdx.x];
  bf16x4 o;
  o[0] = (__bf16)f.x; o[1] = (__bf16)f.y; o[2] = (__bf16)f.z; o[3] = (__bf16)f.w;
  *(bf16x4*)(out + (long long)r * 512 + threadIdx.x * 4) = o;
}

// f32 -> bf16, 4 per thread (n % 4 == 0)
__global__ __launch_bounds__(256) void cvt_f32_bf16(
    const float* __restrict__ src, __bf16* __restrict__ dst, int n)
{
  int i = (blockIdx.x * 256 + threadIdx.x) * 4;
  if (i < n) {
    float4 f = *(const float4*)(src + i);
    bf16x4 o;
    o[0] = (__bf16)f.x; o[1] = (__bf16)f.y; o[2] = (__bf16)f.z; o[3] = (__bf16)f.w;
    *(bf16x4*)(dst + i) = o;
  }
}

__global__ void zero_f32(float* __restrict__ p, int n) {
  int i = blockIdx.x * blockDim.x + threadIdx.x;
  if (i < n) p[i] = 0.0f;
}

// ---------------------------------------------------------------------------
// Per-row masked NLL over 32000 logits; atomics into acc[2] = {sum, count}.
// ---------------------------------------------------------------------------
__global__ __launch_bounds__(256) void nll_rows(
    const float* __restrict__ logits, const int* __restrict__ y,
    float* __restrict__ acc)
{
  __shared__ float red[256];
  const int row = blockIdx.x;
  const int b = row / 63, t = row - b * 63;
  const int tgt = y[b * 64 + t + 1];
  const float* lr = logits + (long long)row * 32000;

  float mx = -3.4e38f;
  for (int i = threadIdx.x; i < 32000; i += 256) mx = fmaxf(mx, lr[i]);
  red[threadIdx.x] = mx; __syncthreads();
  for (int s = 128; s > 0; s >>= 1) {
    if (threadIdx.x < s) red[threadIdx.x] = fmaxf(red[threadIdx.x], red[threadIdx.x + s]);
    __syncthreads();
  }
  mx = red[0]; __syncthreads();

  float sm = 0.0f;
  for (int i = threadIdx.x; i < 32000; i += 256) sm += __expf(lr[i] - mx);
  red[threadIdx.x] = sm; __syncthreads();
  for (int s = 128; s > 0; s >>= 1) {
    if (threadIdx.x < s) red[threadIdx.x] += red[threadIdx.x + s];
    __syncthreads();
  }
  if (threadIdx.x == 0 && tgt != 0) {
    float nll = logf(red[0]) + mx - lr[tgt];
    atomicAdd(&acc[0], nll);
    atomicAdd(&acc[1], 1.0f);
  }
}

__global__ void finalize_loss(const float* __restrict__ acc, float* __restrict__ out) {
  out[0] = acc[0] / fmaxf(acc[1], 1.0f);
}

// ---------------------------------------------------------------------------
// Host orchestration
// ---------------------------------------------------------------------------
extern "C" void kernel_launch(void* const* d_in, const int* in_sizes, int n_in,
                              void* d_out, int out_size, void* d_ws, size_t ws_size,
                              hipStream_t stream) {
  const int B = 64, S = 96, Tt = 64, V = 32000, D = 512;
  const int Td = Tt - 1;

  const int*   x       = (const int*)  d_in[0];
  const int*   y       = (const int*)  d_in[1];
  const float* src_emb = (const float*)d_in[2];
  const float* tgt_emb = (const float*)d_in[3];
  const float* f32_Wih[2][2] = {{(const float*)d_in[4],  (const float*)d_in[7]},
                                {(const float*)d_in[10], (const float*)d_in[13]}};
  const float* f32_Whh[2][2] = {{(const float*)d_in[5],  (const float*)d_in[8]},
                                {(const float*)d_in[11], (const float*)d_in[14]}};
  const float* ebias[2][2]   = {{(const float*)d_in[6],  (const float*)d_in[9]},
                                {(const float*)d_in[12], (const float*)d_in[15]}};
  const float* br_hW   = (const float*)d_in[16];
  const float* br_hb   = (const float*)d_in[17];
  const float* br_cW   = (const float*)d_in[18];
  const float* br_cb   = (const float*)d_in[19];
  const float* dec_Wih = (const float*)d_in[20];
  const float* dec_Whh = (const float*)d_in[21];
  const float* dec_b   = (const float*)d_in[22];
  const float* out_W   = (const float*)d_in[23];
  const float* out_b   = (const float*)d_in[24];

  // ---- workspace carve (byte-based, 256B aligned chunks) ----
  char* base = (char*)d_ws;
  size_t off = 0;
  auto carve = [&](size_t bytes) -> char* {
    char* p = base + off;
    off += (bytes + 255) & ~(size_t)255;
    return p;
  };
  float*  xg       = (float*)carve((size_t)B * S * 4 * D * 4);
  float*  ecb[2]; for (int l = 0; l < 2; ++l) ecb[l] = (float*)carve((size_t)B * 2 * D * 4);
  float*  dcb[2]; for (int l = 0; l < 2; ++l) dcb[l] = (float*)carve((size_t)B * D * 4);
  float*  lacc     = (float*)carve(2 * 4);
  float*  brtmpf   = (float*)carve((size_t)B * D * 4);
  __bf16* enc_in   = (__bf16*)carve((size_t)B * S * D * 2);        // reuse: dec_in
  __bf16* l0out    = (__bf16*)carve((size_t)B * S * 2 * D * 2);    // reuse: dec1+dec2
  __bf16* ehb[2][2]; __bf16* dhb[2][2];
  for (int l = 0; l < 2; ++l) {
    ehb[l][0] = (__bf16*)carve((size_t)B * 2 * D * 2);
    ehb[l][1] = (__bf16*)carve((size_t)B * 2 * D * 2);
  }
  for (int l = 0; l < 2; ++l) {
    dhb[l][0] = (__bf16*)carve((size_t)B * D * 2);
    dhb[l][1] = (__bf16*)carve((size_t)B * D * 2);
  }
  __bf16* brtmpb   = (__bf16*)carve((size_t)B * 2 * D * 2);
  __bf16* wWih[2][2]; __bf16* wWhh[2][2];
  for (int l = 0; l < 2; ++l)
    for (int d = 0; d < 2; ++d) {
      wWih[l][d] = (__bf16*)carve((size_t)4 * D * (l ? 2 * D : D) * 2);
      wWhh[l][d] = (__bf16*)carve((size_t)4 * D * D * 2);
    }
  __bf16* wdWih = (__bf16*)carve((size_t)2 * 4 * D * D * 2);
  __bf16* wdWhh = (__bf16*)carve((size_t)2 * 4 * D * D * 2);
  __bf16* wbrh  = (__bf16*)carve((size_t)D * 2 * D * 2);
  __bf16* wbrc  = (__bf16*)carve((size_t)D * 2 * D * 2);
  __bf16* wout  = (__bf16*)carve((size_t)V * D * 2);
  __bf16* dec_in = enc_in;
  __bf16* dec1   = l0out;
  __bf16* dec2   = l0out + (size_t)B * Td * D;

  float* logits = (float*)d_out;

  auto gemm = [&](const __bf16* A, int lda, const __bf16* W, int ldw,
                  const float* bias, float* Y, int ldy, int M, int N, int K) {
    dim3 g(N / 128, M / 64);
    gemm_bf16_wmma<<<g, 256, 0, stream>>>(A, lda, W, ldw, bias, Y, ldy, K);
  };
  auto cvt = [&](const float* s, __bf16* d, int n) {
    cvt_f32_bf16<<<(n / 4 + 255) / 256, 256, 0, stream>>>(s, d, n);
  };
  auto zero = [&](void* p, int nf32) {
    zero_f32<<<(nf32 + 255) / 256, 256, 0, stream>>>((float*)p, nf32);
  };

  // ---- pre-convert all GEMM weights to bf16 (L2-resident afterwards) ----
  for (int l = 0; l < 2; ++l)
    for (int d = 0; d < 2; ++d) {
      cvt(f32_Wih[l][d], wWih[l][d], 4 * D * (l ? 2 * D : D));
      cvt(f32_Whh[l][d], wWhh[l][d], 4 * D * D);
    }
  cvt(dec_Wih, wdWih, 2 * 4 * D * D);
  cvt(dec_Whh, wdWhh, 2 * 4 * D * D);
  cvt(br_hW, wbrh, D * 2 * D);
  cvt(br_cW, wbrc, D * 2 * D);
  cvt(out_W, wout, V * D);

  // ---------------- encoder ----------------
  embed_gather_bf16<<<B * S, 128, 0, stream>>>(x, S, src_emb, enc_in, S);

  for (int l = 0; l < 2; ++l) {
    const __bf16* Ain = (l == 0) ? enc_in : l0out;
    const int Kin = (l == 0) ? D : 2 * D;
    zero(ehb[l][0], B * 2 * D / 2);
    zero(ehb[l][1], B * 2 * D / 2);
    zero(ecb[l],    B * 2 * D);
    for (int dir = 0; dir < 2; ++dir) {
      gemm(Ain, Kin, wWih[l][dir], Kin, ebias[l][dir], xg, 4 * D, B * S, 4 * D, Kin);
      const int coff = dir * D;
      int cur = 0;
      for (int t = 0; t < S; ++t) {
        const int s = dir ? (S - 1 - t) : t;
        __bf16* yst = (l == 0) ? (l0out + (size_t)s * 2 * D + coff) : nullptr;
        lstm_step_wmma<<<8, 256, 0, stream>>>(
            xg + (size_t)s * 4 * D, (long long)S * 4 * D,
            wWhh[l][dir],
            ehb[l][cur] + coff, 2 * D,
            ehb[l][cur ^ 1] + coff, 2 * D,
            ecb[l] + coff, 2 * D,
            yst, (long long)S * 2 * D);
        cur ^= 1;
      }
      // S even -> final state in ehb[l][0]
    }
  }

  // ---------------- bridge ----------------
  for (int l = 0; l < 2; ++l) {
    cvt(ecb[l], brtmpb, B * 2 * D);
    gemm(brtmpb, 2 * D, wbrc, 2 * D, br_cb, dcb[l], D, B, D, 2 * D);
    gemm(ehb[l][0], 2 * D, wbrh, 2 * D, br_hb, brtmpf, D, B, D, 2 * D);
    cvt(brtmpf, dhb[l][0], B * D);
  }

  // ---------------- decoder ----------------
  embed_gather_bf16<<<B * Td, 128, 0, stream>>>(y, Tt, tgt_emb, dec_in, Td);

  for (int l = 0; l < 2; ++l) {
    const __bf16* Ain = (l == 0) ? dec_in : dec1;
    __bf16* ybuf = (l == 0) ? dec1 : dec2;
    gemm(Ain, D, wdWih + (size_t)l * 4 * D * D, D, dec_b + (size_t)l * 4 * D,
         xg, 4 * D, B * Td, 4 * D, D);
    int cur = 0;
    for (int t = 0; t < Td; ++t) {
      lstm_step_wmma<<<8, 256, 0, stream>>>(
          xg + (size_t)t * 4 * D, (long long)Td * 4 * D,
          wdWhh + (size_t)l * 4 * D * D,
          dhb[l][cur], D,
          dhb[l][cur ^ 1], D,
          dcb[l], D,
          ybuf + (size_t)t * D, (long long)Td * D);
      cur ^= 1;
    }
  }

  // ---------------- logits + loss ----------------
  gemm(dec2, D, wout, D, out_b, logits, V, B * Td, V, D);

  zero(lacc, 2);
  nll_rows<<<B * Td, 256, 0, stream>>>(logits, y, lacc);
  finalize_loss<<<1, 1, 0, stream>>>(lacc, logits + (out_size - 1));
}
